// MLP_RNN_Classifier_6648609374923
// MI455X (gfx1250) — matrix operand order repaired and checked by reading, not compiled, not run on hardware
//
#include <hip/hip_runtime.h>
#include <hip/hip_bf16.h>

#define B_ 64
#define T_ 128
#define H_ 512
#define C_ 10

typedef __bf16 bf16_t;
typedef __bf16 v16bf __attribute__((ext_vector_type(16)));
typedef float  v8f   __attribute__((ext_vector_type(8)));
typedef unsigned int u32x4 __attribute__((ext_vector_type(4)));

union Frag { v16bf v; u32x4 u[2]; };

// ---------------------------------------------------------------------------
// Kernel 1: convert x -> bf16, split W_i2h into bf16 W_x (k<512) and W_h (k>=512)
// ---------------------------------------------------------------------------
__global__ void prep_convert(const float* __restrict__ x,
                             const float* __restrict__ W_i2h,
                             bf16_t* __restrict__ xbf,
                             bf16_t* __restrict__ Wxbf,
                             bf16_t* __restrict__ Whbf) {
    const long NX = (long)B_ * T_ * H_;       // 4,194,304
    const long NW = (long)H_ * H_;            // 262,144
    long stride = (long)gridDim.x * blockDim.x;
    for (long i = (long)blockIdx.x * blockDim.x + threadIdx.x; i < NX; i += stride)
        xbf[i] = (bf16_t)x[i];
    for (long i = (long)blockIdx.x * blockDim.x + threadIdx.x; i < NW; i += stride) {
        int n = (int)(i >> 9), k = (int)(i & (H_ - 1));
        Wxbf[i] = (bf16_t)W_i2h[n * 2 * H_ + k];
        Whbf[i] = (bf16_t)W_i2h[n * 2 * H_ + H_ + k];
    }
}

// ---------------------------------------------------------------------------
// Kernel 2: Xproj[t*B+b][n] = bf16( x[b,t,:] @ W_x.T + b_i2h )  (no recurrence)
// One 16x16 tile per wave; K=512 in 16 chunks of 32 (bf16 WMMA, f32 accum).
// Output layout [T][B][H] so the serial loop reads one contiguous 16KB tile.
// ---------------------------------------------------------------------------
__global__ void __launch_bounds__(256) xproj_kernel(
        const bf16_t* __restrict__ xbf,
        const bf16_t* __restrict__ Wxbf,
        const float*  __restrict__ b_i2h,
        bf16_t* __restrict__ Xproj) {
    int wave = threadIdx.x >> 5;
    int lane = threadIdx.x & 31;
    int half = lane >> 4;
    int l16  = lane & 15;
    int tile    = blockIdx.x * 8 + wave;     // 16384 tiles total
    int rowTile = tile >> 5;                 // 512 row tiles over (b,t)
    int colTile = tile & 31;                 // 32 col tiles over n
    int rowBase = rowTile * 16;
    int n0      = colTile * 16;

    const bf16_t* aBase = xbf  + (long)(rowBase + l16) * H_;  // A row = b*T+t row
    const bf16_t* bBase = Wxbf + (long)(n0 + l16) * H_;       // B col n = weight row

    v8f acc = {};
#pragma unroll 4
    for (int kc = 0; kc < H_ / 32; ++kc) {
        Frag a, b;
        int ka = kc * 32 + half * 16;    // A: 16 contiguous k per lane
        a.u[0] = *(const u32x4*)(aBase + ka);
        a.u[1] = *(const u32x4*)(aBase + ka + 8);
        int kb = kc * 32 + half * 16;    // B: 16 contiguous k per lane
        b.u[0] = *(const u32x4*)(bBase + kb);
        b.u[1] = *(const u32x4*)(bBase + kb + 8);
        acc = __builtin_amdgcn_wmma_f32_16x16x32_bf16(false, a.v, false, b.v,
                                                      (short)0, acc, false, false);
    }
    float bias = b_i2h[n0 + l16];
#pragma unroll
    for (int r = 0; r < 8; ++r) {
        int M = r + half * 8;
        int i = rowBase + M;               // i = b*T + t
        int bb = i >> 7;                   // /T_
        int t  = i & (T_ - 1);
        Xproj[(long)(t * B_ + bb) * H_ + n0 + l16] = (bf16_t)(acc[r] + bias);
    }
}

// ---------------------------------------------------------------------------
// Async prefetch of one contiguous 16KB Xproj tile into LDS (ASYNCcnt path).
// 512 threads x 2 x b128 per call.
// ---------------------------------------------------------------------------
__device__ __forceinline__ void async_tile_prefetch(const bf16_t* gsrc_tile,
                                                    bf16_t* lds_tile, int tid) {
    const char* s0 = (const char*)gsrc_tile + tid * 16;
    const char* s1 = s0 + 8192;
    unsigned d0 = (unsigned)(uintptr_t)lds_tile + tid * 16;
    unsigned d1 = d0 + 8192;
    asm volatile("global_load_async_to_lds_b128 %0, %1, off"
                 :: "v"(d0), "v"(s0) : "memory");
    asm volatile("global_load_async_to_lds_b128 %0, %1, off"
                 :: "v"(d1), "v"(s1) : "memory");
}

// ---------------------------------------------------------------------------
// Kernel 3: persistent RNN recurrence. 4 blocks x 512 threads (16 waves).
// Each block owns 16 batch rows; each wave owns a 32-column slice of h.
// Per step: acc = LDS-prefetched Xproj tile, then += h @ W_h.T (bf16 WMMA).
// Next step's Xproj tile streams in via async global->LDS during the WMMAs.
// ---------------------------------------------------------------------------
__global__ void __launch_bounds__(512) rnn_kernel(
        const bf16_t* __restrict__ Xproj,
        const bf16_t* __restrict__ Whbf,
        const float*  __restrict__ x,
        const float*  __restrict__ W_i2o,
        const float*  __restrict__ b_i2o,
        float* __restrict__ logitsOut) {
    __shared__ __align__(16) bf16_t hbuf[16][H_ + 8];   // +8 bf16 pad vs bank conflicts
    __shared__ __align__(16) bf16_t xpbuf[2][16][H_];   // double-buffered Xproj tile

    int tid  = threadIdx.x;
    int wave = tid >> 5, lane = tid & 31;
    int half = lane >> 4, l16 = lane & 15;
    int b0   = blockIdx.x * 16;

    // h0 = 0
    for (int i = tid; i < 16 * H_; i += 512)
        hbuf[i >> 9][i & (H_ - 1)] = (bf16_t)0.0f;

    // prefetch Xproj tile for t = 0
    async_tile_prefetch(Xproj + ((long)0 * B_ + b0) * H_, &xpbuf[0][0][0], tid);

    int n0a = wave * 32;
    int n0b = n0a + 16;
    const bf16_t* wA = Whbf + (long)(n0a + l16) * H_;
    const bf16_t* wB = Whbf + (long)(n0b + l16) * H_;

    v8f acc0 = {}, acc1 = {};
    for (int t = 0; t < T_; ++t) {
        asm volatile("s_wait_asynccnt 0x0" ::: "memory");  // tile t landed in LDS
        __syncthreads();                                   // + h writes visible
        int buf = t & 1;

        // init accumulators from the LDS-resident input projection (+bias)
#pragma unroll
        for (int r = 0; r < 8; ++r) {
            int M = r + half * 8;
            acc0[r] = (float)xpbuf[buf][M][n0a + l16];
            acc1[r] = (float)xpbuf[buf][M][n0b + l16];
        }

        // prefetch next step's tile while this step's WMMA chain runs
        if (t + 1 < T_)
            async_tile_prefetch(Xproj + ((long)(t + 1) * B_ + b0) * H_,
                                &xpbuf[buf ^ 1][0][0], tid);

        // accumulate h @ W_h.T  (A frag shared by both column tiles)
#pragma unroll 4
        for (int kc = 0; kc < H_ / 32; ++kc) {
            Frag a, bA, bB;
            int ka = kc * 32 + half * 16;
            a.u[0] = *(const u32x4*)(&hbuf[l16][ka]);
            a.u[1] = *(const u32x4*)(&hbuf[l16][ka + 8]);
            int kb = kc * 32 + half * 16;
            bA.u[0] = *(const u32x4*)(wA + kb);
            bA.u[1] = *(const u32x4*)(wA + kb + 8);
            bB.u[0] = *(const u32x4*)(wB + kb);
            bB.u[1] = *(const u32x4*)(wB + kb + 8);
            acc0 = __builtin_amdgcn_wmma_f32_16x16x32_bf16(false, a.v, false, bA.v,
                                                           (short)0, acc0, false, false);
            acc1 = __builtin_amdgcn_wmma_f32_16x16x32_bf16(false, a.v, false, bB.v,
                                                           (short)0, acc1, false, false);
        }
        if (t == T_ - 1) break;            // last h' is never consumed
        __syncthreads();                   // all reads of old h done
#pragma unroll
        for (int r = 0; r < 8; ++r) {
            int M = r + half * 8;
            hbuf[M][n0a + l16] = (bf16_t)acc0[r];
            hbuf[M][n0b + l16] = (bf16_t)acc1[r];
        }
    }
    __syncthreads();

    // logits at t = T-1 use combined = [x[:,T-1,:], h_{T-2}] (hbuf, pre-update)
    if (tid < 16 * C_) {
        int bl = tid / C_, c = tid % C_;
        int bg = b0 + bl;
        float s = b_i2o[c];
        const float* xr = x + ((long)bg * T_ + (T_ - 1)) * H_;
        const float* wr = W_i2o + (long)c * 2 * H_;
        for (int k = 0; k < H_; ++k) s += xr[k] * wr[k];
        for (int k = 0; k < H_; ++k) s += (float)hbuf[bl][k] * wr[H_ + k];
        logitsOut[bg * C_ + c] = s;
    }
}

// ---------------------------------------------------------------------------
// Kernel 4: log_softmax + NLL loss + accuracy. 1 block, 64 threads (1 per row).
// d_out layout: outputs[64*10], loss, acc
// ---------------------------------------------------------------------------
__global__ void finalize_kernel(const float* __restrict__ logits,
                                const int*   __restrict__ y,
                                float* __restrict__ out) {
    __shared__ float ls[64], as[64];
    int b = threadIdx.x;
    float l[C_];
    float m = -1e30f;
    for (int c = 0; c < C_; ++c) { l[c] = logits[b * C_ + c]; m = fmaxf(m, l[c]); }
    float s = 0.f;
    for (int c = 0; c < C_; ++c) s += expf(l[c] - m);
    float logZ = m + logf(s);
    int pred = 0; float best = -1e30f;
    for (int c = 0; c < C_; ++c) {
        float o = l[c] - logZ;
        out[b * C_ + c] = o;
        if (o > best) { best = o; pred = c; }
    }
    int yy = y[b];
    ls[b] = -(l[yy] - logZ);
    as[b] = (pred == yy) ? 1.0f : 0.0f;
    __syncthreads();
    for (int off = 32; off >= 1; off >>= 1) {
        if (b < off) { ls[b] += ls[b + off]; as[b] += as[b + off]; }
        __syncthreads();
    }
    if (b == 0) {
        out[B_ * C_]     = ls[0] / B_;
        out[B_ * C_ + 1] = as[0] / B_;
    }
}

// ---------------------------------------------------------------------------
extern "C" void kernel_launch(void* const* d_in, const int* in_sizes, int n_in,
                              void* d_out, int out_size, void* d_ws, size_t ws_size,
                              hipStream_t stream) {
    (void)in_sizes; (void)n_in; (void)out_size; (void)ws_size;
    const float* x     = (const float*)d_in[0];
    const int*   y     = (const int*)  d_in[1];
    const float* W_i2h = (const float*)d_in[2];
    const float* b_i2h = (const float*)d_in[3];
    const float* W_i2o = (const float*)d_in[4];
    const float* b_i2o = (const float*)d_in[5];
    float* out = (float*)d_out;

    char* ws = (char*)d_ws;
    // workspace layout (~17 MB total)
    bf16_t* xbf    = (bf16_t*)(ws);                                  //  8 MB
    bf16_t* Wxbf   = (bf16_t*)(ws + (size_t)8  * 1024 * 1024);       // 512 KB
    bf16_t* Whbf   = (bf16_t*)(ws + (size_t)8  * 1024 * 1024 + 512 * 1024);
    bf16_t* Xproj  = (bf16_t*)(ws + (size_t)9  * 1024 * 1024);       //  8 MB, [T][B][H]
    float*  logits = (float*) (ws + (size_t)17 * 1024 * 1024);       // 64*10 f32

    prep_convert   <<<2048, 256, 0, stream>>>(x, W_i2h, xbf, Wxbf, Whbf);
    xproj_kernel   <<<2048, 256, 0, stream>>>(xbf, Wxbf, b_i2h, Xproj);
    rnn_kernel     <<<4,    512, 0, stream>>>(Xproj, Whbf, x, W_i2o, b_i2o, logits);
    finalize_kernel<<<1,     64, 0, stream>>>(logits, y, out);
}
